// liGRU_317827580235
// MI455X (gfx1250) — matrix operand order repaired and checked by reading, not compiled, not run on hardware
//
#include <hip/hip_runtime.h>
#include <hip/hip_bf16.h>

// liGRU (2 layers) for MI455X / gfx1250, wave32 + WMMA.
// T=1500 B=16 D=512 H=1024. B==16 -> each timestep is one WMMA M-tile.

#define T_STEPS 1500
#define BATCH   16
#define DIN0    512
#define HID     1024
#define KEEPF   0.8f
#define BN_EPSF 1e-5f
#define NBLK    16          // recurrence blocks = HID / 64

typedef __attribute__((ext_vector_type(16))) _Float16 v16h;
typedef __attribute__((ext_vector_type(8)))  _Float16 v8h;
typedef __attribute__((ext_vector_type(8)))  float    v8f;

static __device__ __forceinline__ v8f wmma16(v16h a, v16h b, v8f c) {
  // v_wmma_f32_16x16x32_f16 : D = A(16x32) x B(32x16) + C(16x16 f32)
  return __builtin_amdgcn_wmma_f32_16x16x32_f16(false, a, false, b, (short)0, c,
                                                false, false);
}

// ---- operand loaders per CDNA5 16-bit VGPR layouts -------------------------
// A (16xK, f16): lanes 0-15 row m, halves 0..7 = K k0..k0+7, halves 8..15 =
// K k0+16..k0+23 ; lanes 16-31 same rows, K k0+8.. / k0+24..
static __device__ __forceinline__ v16h load_a_f32cvt(const float* __restrict__ base,
                                                     int ldk, int k0, int lane) {
  const int m  = lane & 15;
  const int hb = (lane >> 4) << 3;
  const float* p = base + (size_t)m * ldk + k0 + hb;
  v8f lo = *(const v8f*)(p);
  v8f hi = *(const v8f*)(p + 16);
  v16h r;
#pragma unroll
  for (int j = 0; j < 8; ++j) { r[j] = (_Float16)lo[j]; r[j + 8] = (_Float16)hi[j]; }
  return r;
}

static __device__ __forceinline__ v16h load_a_f16(const _Float16* __restrict__ base,
                                                  int ldk, int k0, int lane) {
  const int m  = lane & 15;
  const int hb = (lane >> 4) << 3;
  const _Float16* p = base + (size_t)m * ldk + k0 + hb;
  v8h lo = *(const v8h*)(p);
  v8h hi = *(const v8h*)(p + 16);
  v16h r;
#pragma unroll
  for (int j = 0; j < 8; ++j) { r[j] = lo[j]; r[j + 8] = hi[j]; }
  return r;
}

// B (Kx16, f16). B[k][n] = W[n][k] (row-major W[N][K]) so column n of B is a
// contiguous row of W. Same K order as A: lanes 0-15 K k0..k0+7 / k0+16..k0+23,
// lanes 16-31 K k0+8..k0+15 / k0+24..k0+31.
static __device__ __forceinline__ v16h load_b_f16(const _Float16* __restrict__ base,
                                                  int ldk, int k0, int lane) {
  const int n    = lane & 15;
  const int koff = (lane >> 4) << 3;
  const _Float16* p = base + (size_t)n * ldk + k0 + koff;
  v8h lo = *(const v8h*)(p);
  v8h hi = *(const v8h*)(p + 16);
  v16h r;
#pragma unroll
  for (int j = 0; j < 8; ++j) { r[j] = lo[j]; r[j + 8] = hi[j]; }
  return r;
}

// ---------------------------------------------------------------------------
// Phase 1: wh = BN(x @ WhW^T), wz = BN(x @ WzW^T). One block per (t, 64-col
// slab); 8 waves = {Wh,Wz} x 4 n-tiles; 4-way K-split accumulators.
template <int DIN>
__global__ void __launch_bounds__(256)
ligru_gemm_bn(const float* __restrict__ X,
              const _Float16* __restrict__ Wh, const _Float16* __restrict__ Wz,
              const float* __restrict__ gh, const float* __restrict__ bh,
              const float* __restrict__ mh, const float* __restrict__ vh,
              const float* __restrict__ gz, const float* __restrict__ bz,
              const float* __restrict__ mz, const float* __restrict__ vz,
              float* __restrict__ whbn, float* __restrict__ wzbn) {
  const int lane = threadIdx.x & 31;
  const int wv   = threadIdx.x >> 5;   // 0..7
  const int mat  = wv >> 2;            // 0 = Wh, 1 = Wz
  const int tile = wv & 3;
  const int t    = blockIdx.x;
  const int n0   = blockIdx.y * 64 + tile * 16;

  const float*    Xb = X + (size_t)t * BATCH * DIN;
  const _Float16* Wb = (mat ? Wz : Wh) + (size_t)n0 * DIN;

  v8f acc0 = {}, acc1 = {}, acc2 = {}, acc3 = {};
  for (int c = 0; c < DIN / 32; c += 4) {
    v16h a0 = load_a_f32cvt(Xb, DIN, (c + 0) * 32, lane);
    v16h b0 = load_b_f16(Wb, DIN, (c + 0) * 32, lane);
    acc0 = wmma16(a0, b0, acc0);
    v16h a1 = load_a_f32cvt(Xb, DIN, (c + 1) * 32, lane);
    v16h b1 = load_b_f16(Wb, DIN, (c + 1) * 32, lane);
    acc1 = wmma16(a1, b1, acc1);
    v16h a2 = load_a_f32cvt(Xb, DIN, (c + 2) * 32, lane);
    v16h b2 = load_b_f16(Wb, DIN, (c + 2) * 32, lane);
    acc2 = wmma16(a2, b2, acc2);
    v16h a3 = load_a_f32cvt(Xb, DIN, (c + 3) * 32, lane);
    v16h b3 = load_b_f16(Wb, DIN, (c + 3) * 32, lane);
    acc3 = wmma16(a3, b3, acc3);
  }
  v8f y = (acc0 + acc1) + (acc2 + acc3);

  const int n  = n0 + (lane & 15);
  const int m0 = (lane >> 4) << 3;
  const float g  = mat ? gz[n] : gh[n];
  const float bb = mat ? bz[n] : bh[n];
  const float mm = mat ? mz[n] : mh[n];
  const float vv = mat ? vz[n] : vh[n];
  const float sc = g * rsqrtf(vv + BN_EPSF);
  const float sh = bb - mm * sc;
  float* out = mat ? wzbn : whbn;
#pragma unroll
  for (int r = 0; r < 8; ++r)
    out[((size_t)t * BATCH + (m0 + r)) * HID + n] = y[r] * sc + sh;
}

// ---------------------------------------------------------------------------
// Phase 2: persistent recurrence. 16 blocks x 8 waves. Block owns 64 hidden
// cols; Uz/Uh slices live in LDS (256 KB, CDNA5 WGP LDS). Grid-wide sync via
// monotonic atomic counter each step. h kept fp32 (master) + f16 (A operand).
__global__ void __launch_bounds__(256)
ligru_recur(const float* __restrict__ whbn, const float* __restrict__ wzbn,
            const _Float16* __restrict__ Uz, const _Float16* __restrict__ Uh,
            float* __restrict__ hf32, _Float16* __restrict__ hf16,
            unsigned* __restrict__ bar, float* __restrict__ out) {
  extern __shared__ char smem[];
  _Float16* ldsU    = (_Float16*)smem;                 // 8 tilecols * 32 chunks * 512 halves
  float*    a_stage = (float*)(smem + 8 * 32 * 512 * 2);

  const int lane = threadIdx.x & 31;
  const int wv   = threadIdx.x >> 5;   // 0..7
  const int mat  = wv >> 2;            // 0 = Uz, 1 = Uh
  const int tile = wv & 3;
  const int nb   = blockIdx.x * 64 + tile * 16;

  // Preload this wave's U slice into LDS, already in B-operand lane order.
  const _Float16* Ub   = (mat ? Uh : Uz) + (size_t)nb * HID;
  _Float16*       myU  = ldsU + (size_t)wv * 32 * 512 + (size_t)lane * 16;
#pragma unroll 4
  for (int c = 0; c < 32; ++c)
    *(v16h*)(myU + (size_t)c * 512) = load_b_f16(Ub, HID, c * 32, lane);
  __syncthreads();

  for (int t = 0; t < T_STEPS; ++t) {
    const _Float16* hA = hf16 + (size_t)(t & 1) * BATCH * HID;
    v8f acc0 = {}, acc1 = {}, acc2 = {}, acc3 = {};
#pragma unroll 8
    for (int c = 0; c < 32; ++c) {
      v16h a = load_a_f16(hA, HID, c * 32, lane);
      v16h b = *(const v16h*)(myU + (size_t)c * 512);
      switch (c & 3) {
        case 0: acc0 = wmma16(a, b, acc0); break;
        case 1: acc1 = wmma16(a, b, acc1); break;
        case 2: acc2 = wmma16(a, b, acc2); break;
        default: acc3 = wmma16(a, b, acc3); break;
      }
    }
    v8f a = (acc0 + acc1) + (acc2 + acc3);
    float* st = a_stage + wv * 256;
#pragma unroll
    for (int r = 0; r < 8; ++r) st[r * 32 + lane] = a[r];
    __syncthreads();

    if (wv < 4) {  // gate math: tile 'wv', combine Uz (wave wv) + Uh (wave wv+4)
      const float* az = a_stage + wv * 256;
      const float* ah = a_stage + (wv + 4) * 256;
      const int n  = blockIdx.x * 64 + wv * 16 + (lane & 15);
      const int m0 = (lane >> 4) << 3;
      const float*    hp32 = hf32 + (size_t)(t & 1) * BATCH * HID;
      float*          hn32 = hf32 + (size_t)((t + 1) & 1) * BATCH * HID;
      _Float16*       hn16 = hf16 + (size_t)((t + 1) & 1) * BATCH * HID;
      if (t + 1 < T_STEPS) {  // prefetch next step's projections (global_prefetch_b8)
        __builtin_prefetch(&wzbn[((size_t)(t + 1) * BATCH + m0) * HID + n], 0, 3);
        __builtin_prefetch(&whbn[((size_t)(t + 1) * BATCH + m0) * HID + n], 0, 3);
      }
#pragma unroll
      for (int r = 0; r < 8; ++r) {
        const int m = m0 + r;
        const size_t gi = ((size_t)t * BATCH + m) * HID + n;
        const float wz = wzbn[gi];
        const float wh = whbn[gi];
        const float hp = hp32[m * HID + n];
        const float z  = 1.0f / (1.0f + __expf(-(wz + az[r * 32 + lane])));
        const float at = wh + ah[r * 32 + lane];
        const float hc = fmaxf(at, 0.0f) * KEEPF;
        const float hn = z * hp + (1.0f - z) * hc;
        hn32[m * HID + n] = hn;
        hn16[m * HID + n] = (_Float16)hn;
        out[gi] = hn;
      }
    }
    __threadfence();
    __syncthreads();  // a_stage consumed; h stores fenced
    if (threadIdx.x == 0) {
      atomicAdd(bar, 1u);
      const unsigned target = (unsigned)NBLK * (unsigned)(t + 1);
      while (*((volatile unsigned*)bar) < target) { }
    }
    __syncthreads();
    __threadfence();
  }
}

// ---------------------------------------------------------------------------
__global__ void ligru_f32_to_f16(const float* __restrict__ s, _Float16* __restrict__ d, int n) {
  int i = blockIdx.x * blockDim.x + threadIdx.x;
  if (i < n) d[i] = (_Float16)s[i];
}

__global__ void ligru_init(float* hf32, _Float16* hf16, unsigned* bar) {
  int i = blockIdx.x * blockDim.x + threadIdx.x;
  if (i < 2 * BATCH * HID) { hf32[i] = 0.0f; hf16[i] = (_Float16)0.0f; }
  if (i < 8) bar[i] = 0u;
}

__global__ void ligru_xlen(const int* __restrict__ xlen, int* __restrict__ dst) {
  if (threadIdx.x < BATCH) dst[threadIdx.x] = xlen[threadIdx.x];
}

// ---------------------------------------------------------------------------
extern "C" void kernel_launch(void* const* d_in, const int* in_sizes, int n_in,
                              void* d_out, int out_size, void* d_ws, size_t ws_size,
                              hipStream_t stream) {
  (void)in_sizes; (void)n_in; (void)out_size; (void)ws_size;

  const float* x    = (const float*)d_in[0];
  const int*   xlen = (const int*)d_in[1];
  const float* whW[2] = { (const float*)d_in[2],  (const float*)d_in[14] };
  const float* wzW[2] = { (const float*)d_in[3],  (const float*)d_in[15] };
  const float* uhW[2] = { (const float*)d_in[4],  (const float*)d_in[16] };
  const float* uzW[2] = { (const float*)d_in[5],  (const float*)d_in[17] };
  const float* bnh[2][4]; const float* bnz[2][4];
  for (int l = 0; l < 2; ++l) {
    const int base = l ? 18 : 6;
    for (int j = 0; j < 4; ++j) {
      bnh[l][j] = (const float*)d_in[base + j];
      bnz[l][j] = (const float*)d_in[base + 4 + j];
    }
  }

  // workspace carve-up
  char* w = (char*)d_ws;
  size_t off = 0;
  float* whbn = (float*)(w + off); off += (size_t)T_STEPS * BATCH * HID * 4;
  float* wzbn = (float*)(w + off); off += (size_t)T_STEPS * BATCH * HID * 4;
  _Float16 *wh16[2], *wz16[2], *uh16[2], *uz16[2];
  wh16[0] = (_Float16*)(w + off); off += (size_t)HID * DIN0 * 2;
  wz16[0] = (_Float16*)(w + off); off += (size_t)HID * DIN0 * 2;
  wh16[1] = (_Float16*)(w + off); off += (size_t)HID * HID * 2;
  wz16[1] = (_Float16*)(w + off); off += (size_t)HID * HID * 2;
  uh16[0] = (_Float16*)(w + off); off += (size_t)HID * HID * 2;
  uz16[0] = (_Float16*)(w + off); off += (size_t)HID * HID * 2;
  uh16[1] = (_Float16*)(w + off); off += (size_t)HID * HID * 2;
  uz16[1] = (_Float16*)(w + off); off += (size_t)HID * HID * 2;
  float*    hf32 = (float*)(w + off);    off += (size_t)2 * BATCH * HID * 4;
  _Float16* hf16 = (_Float16*)(w + off); off += (size_t)2 * BATCH * HID * 2;
  unsigned* bar  = (unsigned*)(w + off); off += 256;

  // f32 -> f16 weight copies (every launch; deterministic)
  auto cvt = [&](const float* s, _Float16* d, int n) {
    ligru_f32_to_f16<<<(n + 255) / 256, 256, 0, stream>>>(s, d, n);
  };
  cvt(whW[0], wh16[0], HID * DIN0);
  cvt(wzW[0], wz16[0], HID * DIN0);
  cvt(whW[1], wh16[1], HID * HID);
  cvt(wzW[1], wz16[1], HID * HID);
  cvt(uhW[0], uh16[0], HID * HID);
  cvt(uzW[0], uz16[0], HID * HID);
  cvt(uhW[1], uh16[1], HID * HID);
  cvt(uzW[1], uz16[1], HID * HID);

  const size_t LDSB = (size_t)8 * 32 * 512 * sizeof(_Float16)   // 256 KB U tiles
                    + (size_t)8 * 256 * sizeof(float);          //   8 KB a_stage
  float* out = (float*)d_out;

  for (int l = 0; l < 2; ++l) {
    ligru_init<<<(2 * BATCH * HID + 255) / 256, 256, 0, stream>>>(hf32, hf16, bar);
    const float* Xl = l ? (const float*)out : x;   // layer-0 h staged in d_out
    if (l == 0) {
      ligru_gemm_bn<DIN0><<<dim3(T_STEPS, HID / 64), 256, 0, stream>>>(
          Xl, wh16[0], wz16[0],
          bnh[0][0], bnh[0][1], bnh[0][2], bnh[0][3],
          bnz[0][0], bnz[0][1], bnz[0][2], bnz[0][3], whbn, wzbn);
    } else {
      ligru_gemm_bn<HID><<<dim3(T_STEPS, HID / 64), 256, 0, stream>>>(
          Xl, wh16[1], wz16[1],
          bnh[1][0], bnh[1][1], bnh[1][2], bnh[1][3],
          bnz[1][0], bnz[1][1], bnz[1][2], bnz[1][3], whbn, wzbn);
    }
    ligru_recur<<<NBLK, 256, LDSB, stream>>>(whbn, wzbn, uz16[l], uh16[l],
                                             hf32, hf16, bar, out);
  }
  ligru_xlen<<<1, 64, 0, stream>>>(xlen, ((int*)d_out) + (size_t)T_STEPS * BATCH * HID);
}
